// MLPDecoder_79113297592641
// MI455X (gfx1250) — matrix operand order repaired and checked by reading, not compiled, not run on hardware
//
#include <hip/hip_runtime.h>
#include <hip/hip_bf16.h>

typedef __attribute__((ext_vector_type(16))) _Float16 v16h;
typedef __attribute__((ext_vector_type(8)))  _Float16 v8h;
typedef __attribute__((ext_vector_type(8)))  float    v8f;

#define Dd   128      // feature dim
#define Cc   5        // num classes
#define TM   128      // edges per block
#define NTHR 256      // 8 waves

// LDS byte offsets (all 16B aligned)
#define OFF_W1  0            // [64 tiles][32lane][16] f16 = 65536 B
#define OFF_W2  65536        // [32 tiles][32lane][16] f16 = 32768 B
#define OFF_WP  98304        // [4 tiles][32lane][16] f16 = 4096 B
#define OFF_A   102400       // [128][256] f16 = 65536 B
#define OFF_H1  167936       // [128][128] f16 = 32768 B
#define OFF_H2  200704       // [128][128] f16 = 32768 B
#define SMEM_BYTES 233472

// A fragment (16x32 f16): lane<16 -> row=lane, k elems {0..15};
// lane>=16 -> row=lane-16, k elems {16..31}
__device__ __forceinline__ v16h load_frag_a(const _Float16* __restrict__ base,
                                            int ldk, int lane) {
  const int half = lane >> 4;
  const int row  = lane & 15;
  const _Float16* p = base + row * ldk + half * 16;
  v8h lo = *(const v8h*)(p);
  v8h hi = *(const v8h*)(p + 8);
  v16h a;
#pragma unroll
  for (int i = 0; i < 8; ++i) { a[i] = lo[i]; a[8 + i] = hi[i]; }
  return a;
}

// B fragment: pre-swizzled in LDS, 16 contiguous halves per lane per tile
__device__ __forceinline__ v16h load_frag_b(const _Float16* __restrict__ tiles,
                                            int t, int lane) {
  return *(const v16h*)(tiles + t * 512 + lane * 16);
}

__global__ __launch_bounds__(NTHR, 1)
void MLPDecoder_79113297592641_kernel(const float* __restrict__ ufeat,
                                      const float* __restrict__ ifeat,
                                      const int*  __restrict__ src,
                                      const int*  __restrict__ dst,
                                      const float* __restrict__ W1,
                                      const float* __restrict__ W2,
                                      const float* __restrict__ Wp,
                                      float* __restrict__ out,
                                      int E) {
  extern __shared__ __align__(16) char smem[];
  _Float16* sW1 = (_Float16*)(smem + OFF_W1);
  _Float16* sW2 = (_Float16*)(smem + OFF_W2);
  _Float16* sWp = (_Float16*)(smem + OFF_WP);
  _Float16* sA  = (_Float16*)(smem + OFF_A);
  _Float16* sH1 = (_Float16*)(smem + OFF_H1);
  _Float16* sH2 = (_Float16*)(smem + OFF_H2);

  const int tid  = threadIdx.x;
  const int lane = tid & 31;
  const int wave = tid >> 5;
  const int blockBase = blockIdx.x * TM;

  // ---- Phase 0a: stage W1 [256,128] f32 -> B-fragment-swizzled f16 LDS ----
  // frag elem (kt,nt,lane,i): k = kt*32 + (lane>=16)*16 + i ; n = nt*16 + (lane&15)
  for (int idx = tid; idx < 8 * 8 * 32 * 16; idx += NTHR) {
    const int i  = idx & 15;
    const int ln = (idx >> 4) & 31;
    const int nt = (idx >> 9) & 7;
    const int kt = idx >> 12;
    const int k  = kt * 32 + ((ln >> 4) << 4) + i;
    const int n  = nt * 16 + (ln & 15);
    sW1[idx] = (_Float16)W1[k * Dd + n];
  }
  // ---- Phase 0b: W2 [128,128] ----
  for (int idx = tid; idx < 4 * 8 * 32 * 16; idx += NTHR) {
    const int i  = idx & 15;
    const int ln = (idx >> 4) & 31;
    const int nt = (idx >> 9) & 7;
    const int kt = idx >> 12;
    const int k  = kt * 32 + ((ln >> 4) << 4) + i;
    const int n  = nt * 16 + (ln & 15);
    sW2[idx] = (_Float16)W2[k * Dd + n];
  }
  // ---- Phase 0c: Wp [128,5] zero-padded to [128,16] ----
  for (int idx = tid; idx < 4 * 32 * 16; idx += NTHR) {
    const int i  = idx & 15;
    const int ln = (idx >> 4) & 31;
    const int kt = idx >> 9;
    const int k  = kt * 32 + ((ln >> 4) << 4) + i;
    const int n  = ln & 15;
    sWp[idx] = (n < Cc) ? (_Float16)Wp[k * Cc + n] : (_Float16)0.0f;
  }

  // ---- Phase 1: gather A tile [128 rows][256 cols] as f16 ----
  for (int idx = tid; idx < TM * 64; idx += NTHR) {
    const int r  = idx >> 6;
    const int c4 = idx & 63;
    int e = blockBase + r; if (e >= E) e = E - 1;
    const int k0 = c4 * 4;
    const float* p = (k0 < Dd)
        ? (ufeat + (size_t)src[e] * Dd + k0)
        : (ifeat + (size_t)dst[e] * Dd + (k0 - Dd));
    const float4 v = *(const float4*)p;
    _Float16* d = sA + r * 256 + k0;
    d[0] = (_Float16)v.x; d[1] = (_Float16)v.y;
    d[2] = (_Float16)v.z; d[3] = (_Float16)v.w;
  }

  __syncthreads();

  // Each wave owns M-tile rows [m0, m0+16)
  const int m0   = wave * 16;
  const int half = lane >> 4;
  const int ncol = lane & 15;

  // ---- GEMM1: H1 = relu(A[16,256] @ W1[256,128]) ----
  // Batch A + all 8 B fragments for a kt before the 8 WMMAs: the scheduler
  // hoists the whole ds_load burst and uses descending partial s_wait_dscnt.
  {
    v8f acc[8];
#pragma unroll
    for (int nt = 0; nt < 8; ++nt) acc[nt] = (v8f){};
#pragma unroll
    for (int kt = 0; kt < 8; ++kt) {
      const v16h a = load_frag_a(sA + m0 * 256 + kt * 32, 256, lane);
      v16h bb[8];
#pragma unroll
      for (int nt = 0; nt < 8; ++nt) bb[nt] = load_frag_b(sW1, kt * 8 + nt, lane);
#pragma unroll
      for (int nt = 0; nt < 8; ++nt)
        acc[nt] = __builtin_amdgcn_wmma_f32_16x16x32_f16(false, a, false, bb[nt],
                                                         (short)0, acc[nt],
                                                         false, false);
    }
#pragma unroll
    for (int nt = 0; nt < 8; ++nt)
#pragma unroll
      for (int i = 0; i < 8; ++i) {
        float v = acc[nt][i] > 0.0f ? acc[nt][i] : 0.0f;   // relu
        sH1[(m0 + half * 8 + i) * Dd + nt * 16 + ncol] = (_Float16)v;
      }
  }

  // ---- GEMM2: H2 = H1[16,128] @ W2[128,128] (own-wave LDS RAW: in-order) ----
  {
    v8f acc[8];
#pragma unroll
    for (int nt = 0; nt < 8; ++nt) acc[nt] = (v8f){};
#pragma unroll
    for (int kt = 0; kt < 4; ++kt) {
      const v16h a = load_frag_a(sH1 + m0 * Dd + kt * 32, Dd, lane);
      v16h bb[8];
#pragma unroll
      for (int nt = 0; nt < 8; ++nt) bb[nt] = load_frag_b(sW2, kt * 8 + nt, lane);
#pragma unroll
      for (int nt = 0; nt < 8; ++nt)
        acc[nt] = __builtin_amdgcn_wmma_f32_16x16x32_f16(false, a, false, bb[nt],
                                                         (short)0, acc[nt],
                                                         false, false);
    }
#pragma unroll
    for (int nt = 0; nt < 8; ++nt)
#pragma unroll
      for (int i = 0; i < 8; ++i)
        sH2[(m0 + half * 8 + i) * Dd + nt * 16 + ncol] = (_Float16)acc[nt][i];
  }

  // ---- GEMM3: out = H2[16,128] @ Wp[128,16(pad)] ----
  {
    v8f acc = {};
    v16h aa[4], bb[4];
#pragma unroll
    for (int kt = 0; kt < 4; ++kt) {
      aa[kt] = load_frag_a(sH2 + m0 * Dd + kt * 32, Dd, lane);
      bb[kt] = load_frag_b(sWp, kt, lane);
    }
#pragma unroll
    for (int kt = 0; kt < 4; ++kt)
      acc = __builtin_amdgcn_wmma_f32_16x16x32_f16(false, aa[kt], false, bb[kt],
                                                   (short)0, acc, false, false);
    if (ncol < Cc) {
#pragma unroll
      for (int i = 0; i < 8; ++i) {
        const int e = blockBase + m0 + half * 8 + i;
        if (e < E) out[(size_t)e * Cc + ncol] = acc[i];
      }
    }
  }
}

extern "C" void kernel_launch(void* const* d_in, const int* in_sizes, int n_in,
                              void* d_out, int out_size, void* d_ws, size_t ws_size,
                              hipStream_t stream) {
  const float* ufeat = (const float*)d_in[0];
  const float* ifeat = (const float*)d_in[1];
  const int*   src   = (const int*)d_in[2];
  const int*   dst   = (const int*)d_in[3];
  const float* W1    = (const float*)d_in[4];
  const float* W2    = (const float*)d_in[5];
  const float* Wp    = (const float*)d_in[6];
  float* out = (float*)d_out;
  const int E = in_sizes[2];

  const int grid = (E + TM - 1) / TM;
  MLPDecoder_79113297592641_kernel<<<grid, NTHR, SMEM_BYTES, stream>>>(
      ufeat, ifeat, src, dst, W1, W2, Wp, out, E);
}
